// DCT_1133871366826
// MI455X (gfx1250) — matrix operand order repaired and checked by reading, not compiled, not run on hardware
//
#include <hip/hip_runtime.h>

// DCT-II as a WMMA GEMM:  out[m,c] = sum_q x[m,q] * W[q,c],
//   W[q,c] = cos(pi/N * (q+0.5) * c),  M=2048, N=K=4096.
// Split-bf16 (bf16x3): x = x_hi + x_lo, W = W_hi + W_lo (bf16 each),
// accumulate hi*hi + hi*lo + lo*hi with v_wmma_f32_16x16x32_bf16 (f32 acc).
// Global -> LDS staging uses CDNA5 GLOBAL_LOAD_ASYNC_TO_LDS_B128 with a
// double-buffered pipeline synchronized by ASYNCcnt + workgroup barriers.
// Block tile 128x128 (dynamic LDS, 80 KB of the WGP's 320 KB) to cut L2
// re-read traffic and reach a 1:1 wmma:ds_load ratio per K-step.

#define M_DIM 2048
#define N_DIM 4096

typedef __attribute__((ext_vector_type(16))) __bf16 v16bf;
typedef __attribute__((ext_vector_type(8)))  float  v8f;
typedef __attribute__((__vector_size__(16))) int    vi4;   // matches builtin proto

union Frag { v16bf v; uint4 q[2]; };

// ---- async global->LDS (CDNA5), with builtin/asm fallback ------------------
#if __has_builtin(__builtin_amdgcn_global_load_async_to_lds_b128)
#define HAVE_ASYNC_BUILTIN 1
#endif

static __device__ __forceinline__ void async_b128(const unsigned short* g,
                                                  unsigned short* l) {
#ifdef HAVE_ASYNC_BUILTIN
    // Builtin signature (probed): (v4i AS1*, v4i AS3*, imm offset, imm cpol)
    __builtin_amdgcn_global_load_async_to_lds_b128(
        (__attribute__((address_space(1))) vi4*)g,
        (__attribute__((address_space(3))) vi4*)l, 0, 0);
#else
    unsigned lds_off =
        (unsigned)(unsigned long long)(__attribute__((address_space(3))) unsigned short*)l;
    asm volatile("global_load_async_to_lds_b128 %0, %1, off"
                 :: "v"(lds_off), "v"((unsigned long long)g)
                 : "memory");
#endif
}

#if __has_builtin(__builtin_amdgcn_s_wait_asynccnt)
#define WAIT_ASYNC(n) __builtin_amdgcn_s_wait_asynccnt(n)
#else
#define WAIT_ASYNC(n) asm volatile("s_wait_asynccnt %0" :: "n"(n) : "memory")
#endif

// ---- bf16 helpers ----------------------------------------------------------
static __device__ __forceinline__ unsigned short f32_to_bf16_rne(float f) {
    union { float f; unsigned u; } in;
    in.f = f;
    unsigned u = in.u;
    u += 0x7FFFu + ((u >> 16) & 1u);   // round-to-nearest-even
    return (unsigned short)(u >> 16);
}

static __device__ __forceinline__ float bf16_to_f32(unsigned short h) {
    union { unsigned u; float f; } out;
    out.u = ((unsigned)h) << 16;
    return out.f;
}

// ---------------------------------------------------------------------------
// Split the f32 input into bf16 hi/lo tables.
// ---------------------------------------------------------------------------
__global__ __launch_bounds__(256) void dct_prep_x(const float* __restrict__ x,
                                                  unsigned short* __restrict__ hi,
                                                  unsigned short* __restrict__ lo) {
    const int i = blockIdx.x * 256 + threadIdx.x;
    const float v = x[i];
    const unsigned short h = f32_to_bf16_rne(v);
    hi[i] = h;
    lo[i] = f32_to_bf16_rne(v - bf16_to_f32(h));
}

// ---------------------------------------------------------------------------
// Transposed cosine table W_t[c][q] = cos(pi/N * (q+0.5) * c) as bf16 hi/lo.
// Exact power-of-two phase reduction: (2q+1)*c mod 4N, then cospif.
// ---------------------------------------------------------------------------
__global__ __launch_bounds__(256) void dct_prep_w(unsigned short* __restrict__ hi,
                                                  unsigned short* __restrict__ lo) {
    const int i = blockIdx.x * 256 + threadIdx.x;   // i < N*N = 2^24
    const int c = i >> 12;                          // frequency index k
    const int q = i & (N_DIM - 1);                  // input index n
    const int t = ((2 * q + 1) * c) & (4 * N_DIM - 1);
    const float v = cospif((float)t * (1.0f / (2.0f * (float)N_DIM)));
    const unsigned short h = f32_to_bf16_rne(v);
    hi[i] = h;
    lo[i] = f32_to_bf16_rne(v - bf16_to_f32(h));
}

// ---------------------------------------------------------------------------
// GEMM: 256 threads = 8 wave32. Workgroup tile 128(M) x 128(C), K-step 32.
// Waves 4(M) x 2(C); each wave owns a 32x64 tile = 2x4 WMMA tiles,
// 24 v_wmma_f32_16x16x32_bf16 per wave per K-step (1:1 with ds_load_b128).
// LDS pitch 40 halfwords (80 B): ds_load_b128 fragment reads are
// conflict-free (row stride 20 dwords covers all 64 banks over 16 lanes).
// ---------------------------------------------------------------------------
struct alignas(16) Tile {
    unsigned short As_hi[128][40];
    unsigned short As_lo[128][40];
    unsigned short Ws_hi[128][40];
    unsigned short Ws_lo[128][40];
};   // 40960 B per buffer

static __device__ __forceinline__ void issue_tile(
        const unsigned short* __restrict__ Xhi,
        const unsigned short* __restrict__ Xlo,
        const unsigned short* __restrict__ Whi,
        const unsigned short* __restrict__ Wlo,
        Tile* t, int m_blk, int c_blk, int q0, int ld_row, int ld_q) {
    // 8 async b128 loads per thread: 128x32 (hi+lo) A slab + 128x32 (hi+lo) W slab
#pragma unroll
    for (int i = 0; i < 2; ++i) {
        const int row = ld_row + i * 64;
        const size_t aoff = (size_t)(m_blk + row) * N_DIM + q0 + ld_q;
        async_b128(Xhi + aoff, &t->As_hi[row][ld_q]);
        async_b128(Xlo + aoff, &t->As_lo[row][ld_q]);
        const size_t woff = (size_t)(c_blk + row) * N_DIM + q0 + ld_q;
        async_b128(Whi + woff, &t->Ws_hi[row][ld_q]);
        async_b128(Wlo + woff, &t->Ws_lo[row][ld_q]);
    }
}

static __device__ __forceinline__ void compute_tile(
        const Tile* t, int wm, int wc, int l16, int half, v8f (*acc)[4]) {
    // B fragments (32x16): lane holds col c = l16; elements K = half*16 + e
    Frag bhi[4], blo[4];
#pragma unroll
    for (int ct = 0; ct < 4; ++ct) {
        const unsigned short* bp = &t->Ws_hi[wc + ct * 16 + l16][half * 16];
        bhi[ct].q[0] = ((const uint4*)bp)[0];
        bhi[ct].q[1] = ((const uint4*)bp)[1];
        const unsigned short* bq = &t->Ws_lo[wc + ct * 16 + l16][half * 16];
        blo[ct].q[0] = ((const uint4*)bq)[0];
        blo[ct].q[1] = ((const uint4*)bq)[1];
    }
    // A fragments (16x32): lane holds row m = l16;
    // K = half*16 + e
#pragma unroll
    for (int mt = 0; mt < 2; ++mt) {
        Frag ahi, alo;
        const unsigned short* ap = &t->As_hi[wm + mt * 16 + l16][half * 16];
        ahi.q[0] = ((const uint4*)ap)[0];
        ahi.q[1] = ((const uint4*)ap)[1];
        const unsigned short* aq = &t->As_lo[wm + mt * 16 + l16][half * 16];
        alo.q[0] = ((const uint4*)aq)[0];
        alo.q[1] = ((const uint4*)aq)[1];

#pragma unroll
        for (int ct = 0; ct < 4; ++ct) {
            acc[mt][ct] = __builtin_amdgcn_wmma_f32_16x16x32_bf16(
                false, ahi.v, false, bhi[ct].v, (short)0, acc[mt][ct], false, false);
            acc[mt][ct] = __builtin_amdgcn_wmma_f32_16x16x32_bf16(
                false, ahi.v, false, blo[ct].v, (short)0, acc[mt][ct], false, false);
            acc[mt][ct] = __builtin_amdgcn_wmma_f32_16x16x32_bf16(
                false, alo.v, false, bhi[ct].v, (short)0, acc[mt][ct], false, false);
        }
    }
}

__global__ __launch_bounds__(256) void dct_wmma_gemm(
        const unsigned short* __restrict__ Xhi,
        const unsigned short* __restrict__ Xlo,
        const unsigned short* __restrict__ Whi,
        const unsigned short* __restrict__ Wlo,
        float* __restrict__ out) {
    extern __shared__ Tile tiles[];   // tiles[2]: 81920 B dynamic LDS

    const int tid  = threadIdx.x;
    const int lane = tid & 31;
    const int wave = tid >> 5;

    const int m_blk = blockIdx.y * 128;
    const int c_blk = blockIdx.x * 128;

    const int wm = (wave >> 1) * 32;   // 0,32,64,96
    const int wc = (wave & 1) * 64;    // 0,64

    const int ld_row = tid >> 2;       // 0..63
    const int ld_q   = (tid & 3) * 8;  // 0,8,16,24 (halfwords)

    const int half = lane >> 4;
    const int l16  = lane & 15;

    const v8f zero = {0.f, 0.f, 0.f, 0.f, 0.f, 0.f, 0.f, 0.f};
    v8f acc[2][4] = {{zero, zero, zero, zero}, {zero, zero, zero, zero}};

    constexpr int KSTEPS = N_DIM / 32;   // 128

    // Pipeline prologue: tile 0 in flight.
    issue_tile(Xhi, Xlo, Whi, Wlo, &tiles[0], m_blk, c_blk, 0, ld_row, ld_q);

    for (int k = 0; k < KSTEPS; ++k) {
        if (k + 1 < KSTEPS) {
            // Next tile's buffer was fully consumed in the previous iteration
            // (trailing barrier of iteration k-1 guarantees it).
            issue_tile(Xhi, Xlo, Whi, Wlo, &tiles[(k + 1) & 1],
                       m_blk, c_blk, (k + 1) * 32, ld_row, ld_q);
            // Async loads complete in order: <=8 outstanding means this
            // wave's 8 loads for tile k have landed in LDS.
            WAIT_ASYNC(8);
        } else {
            WAIT_ASYNC(0);
        }
        __syncthreads();   // all waves' tile-k data visible in LDS

        compute_tile(&tiles[k & 1], wm, wc, l16, half, acc);

        __syncthreads();   // everyone done reading buffer (k&1) before reuse
    }

    // Epilogue: C/D layout — VGPR r: lanes 0-15 -> M=r, lanes 16-31 -> M=8+r
#pragma unroll
    for (int mt = 0; mt < 2; ++mt) {
#pragma unroll
        for (int ct = 0; ct < 4; ++ct) {
            const int c = c_blk + wc + ct * 16 + l16;
#pragma unroll
            for (int r = 0; r < 8; ++r) {
                const int m = m_blk + wm + mt * 16 + half * 8 + r;
                out[(size_t)m * N_DIM + c] = acc[mt][ct][r];
            }
        }
    }
}

// ---------------------------------------------------------------------------
// Launch. Workspace layout (bf16/ushort):
//   Xhi: 2048*4096, Xlo: 2048*4096, Whi: 4096*4096, Wlo: 4096*4096  (96 MB)
// Recomputed every call -> deterministic, graph-capture safe.
// ---------------------------------------------------------------------------
extern "C" void kernel_launch(void* const* d_in, const int* in_sizes, int n_in,
                              void* d_out, int out_size, void* d_ws, size_t ws_size,
                              hipStream_t stream) {
    (void)in_sizes; (void)n_in; (void)out_size; (void)ws_size;

    const float* x = (const float*)d_in[0];
    float* out = (float*)d_out;

    unsigned short* xhi = (unsigned short*)d_ws;
    unsigned short* xlo = xhi + (size_t)M_DIM * N_DIM;
    unsigned short* whi = xlo + (size_t)M_DIM * N_DIM;
    unsigned short* wlo = whi + (size_t)N_DIM * N_DIM;

    dct_prep_x<<<(M_DIM * N_DIM) / 256, 256, 0, stream>>>(x, xhi, xlo);
    dct_prep_w<<<(N_DIM * N_DIM) / 256, 256, 0, stream>>>(whi, wlo);

    dim3 grid(N_DIM / 128, M_DIM / 128);
    dct_wmma_gemm<<<grid, 256, 2 * sizeof(Tile), stream>>>(xhi, xlo, whi, wlo, out);
}
